// ModulatedConv2D_27255862461182
// MI455X (gfx1250) — matrix operand order repaired and checked by reading, not compiled, not run on hardware
//
#include <hip/hip_runtime.h>

typedef __attribute__((ext_vector_type(8)))  float  v8f;
typedef __attribute__((ext_vector_type(8)))  __bf16 v8bf;
typedef __attribute__((ext_vector_type(16))) __bf16 v16bf;

#define BATCH 16
#define CH    512
#define HW    64
#define KTOT  (CH*9)        // 4608
#define NKT   (KTOT/32)     // 144 k-tiles (32-wide)
#define NMT   (CH/16)       // 32 m-tiles
#define LSTR  40            // LDS channel stride (halves): 80B, 16B-aligned, bank-spread
#define NSTEP 48            // 3 kh * 16 channel-chunks

// ---------- stage 1: style[b,o] = dot(w[b,:], lw[o,:]) + lb[o] ----------
__global__ void style_kernel(const float* __restrict__ w, const float* __restrict__ lw,
                             const float* __restrict__ lb, float* __restrict__ style) {
  int tid = blockIdx.x * blockDim.x + threadIdx.x;
  if (tid >= BATCH * CH) return;
  int b = tid / CH, o = tid % CH;
  const float4* wr = (const float4*)(w + (size_t)b * CH);
  const float4* lr = (const float4*)(lw + (size_t)o * CH);
  float acc = 0.f;
  for (int d = 0; d < CH / 4; ++d) {
    float4 a = wr[d], c = lr[d];
    acc += a.x * c.x + a.y * c.y + a.z * c.z + a.w * c.w;
  }
  style[tid] = acc + lb[o];
}

// ---------- stage 2: wsq[o,i] = sum over 9 taps of cw^2 ----------
__global__ void wsq_kernel(const float* __restrict__ cw, float* __restrict__ wsq) {
  int tid = blockIdx.x * blockDim.x + threadIdx.x;
  if (tid >= CH * CH) return;
  const float* p = cw + (size_t)tid * 9;
  float s = 0.f;
  #pragma unroll
  for (int k = 0; k < 9; ++k) s += p[k] * p[k];
  wsq[tid] = s;
}

// ---------- stage 3: rdenom[b,o] = rsqrt(sum_i style[b,i]^2 * wsq[o,i] + eps) ----------
__global__ void rdenom_kernel(const float* __restrict__ style, const float* __restrict__ wsq,
                              float* __restrict__ rdenom) {
  int tid = blockIdx.x * blockDim.x + threadIdx.x;
  if (tid >= BATCH * CH) return;
  int b = tid / CH, o = tid % CH;
  const float4* st = (const float4*)(style + (size_t)b * CH);
  const float4* wq = (const float4*)(wsq + (size_t)o * CH);
  float acc = 0.f;
  for (int d = 0; d < CH / 4; ++d) {
    float4 s = st[d], q = wq[d];
    acc += s.x * s.x * q.x + s.y * s.y * q.y + s.z * s.z * q.z + s.w * s.w * q.w;
  }
  rdenom[tid] = rsqrtf(acc + 1e-6f);
}

// ---------- stage 4: pack cw into bf16 WMMA A-fragment layout ----------
// Apack[mt][kt][lane][e], K = tap*512 + i  (tap = kh*3+kw)
// lanes 0-15 : M=lane,    kloc order [0..7, 16..23]
// lanes 16-31: M=lane-16, kloc order [8..15, 24..31]
__global__ void pack_kernel(const float* __restrict__ cw, unsigned short* __restrict__ apack_us) {
  int tid = blockIdx.x * blockDim.x + threadIdx.x;
  if (tid >= NMT * NKT * 32) return;
  int lane = tid & 31;
  int kt   = (tid >> 5) % NKT;
  int mt   = tid / (32 * NKT);
  int o    = mt * 16 + (lane & 15);
  __bf16* dst = ((__bf16*)apack_us) + (size_t)tid * 16;
  #pragma unroll
  for (int e = 0; e < 16; ++e) {
    int kloc = (lane < 16) ? ((e < 8) ? e : e + 8) : ((e < 8) ? e + 8 : e + 16);
    int kg  = kt * 32 + kloc;
    int tap = kg >> 9;     // K = tap*512 + i
    int i   = kg & 511;
    float v = cw[((size_t)o * CH + i) * 9 + tap];
    dst[e] = (__bf16)v;
  }
}

// ---------- main: implicit-GEMM modulated conv, bf16 WMMA, software-pipelined ----------
__global__ __launch_bounds__(256)
void modconv_kernel(const float* __restrict__ x, const float* __restrict__ style,
                    const float* __restrict__ rdenom,
                    const unsigned short* __restrict__ apack_us,
                    float* __restrict__ out) {
  const int y   = blockIdx.x;   // output row, 0..63
  const int mb  = blockIdx.y;   // M-tile of 128, 0..3
  const int b   = blockIdx.z;   // batch
  const int tid = threadIdx.x;
  const int lane = tid & 31, wv = tid >> 5;
  const int mtg   = mb * 8 + wv;      // global 16-row m-tile, 0..31
  const int mbase = mtg * 16;
  const int ncol  = lane & 15;
  const int kbase = (lane >> 4) << 3; // 0 or 8

  __shared__ float sstyle[CH];
  __shared__ __align__(32) __bf16 xl0[66 * LSTR];
  __shared__ __align__(32) __bf16 xl1[66 * LSTR];

  for (int t = tid; t < CH; t += 256) sstyle[t] = style[(size_t)b * CH + t];

  const __bf16* apack = (const __bf16*)apack_us;
  v8f acc[4] = {};
  v16bf afA[3], afB[3];
  float va[4], vb[4], vc[4];

  // step s (0..47): kh = s>>4, channel chunk ic0 = (s&15)*32
  // ---- prefetch global x row (with halo) into registers; pure adds, coalesced b32 ----
  auto fillLoad = [&](int s) {
    const int kh  = s >> 4;
    const int ic0 = (s & 15) << 5;
    const int row = y + kh - 1;
    const bool rowok = (unsigned)row < (unsigned)HW;
    const int rc = rowok ? row : 0;
    #pragma unroll
    for (int j = 0; j < 4; ++j) {
      const int iloc = wv * 4 + j;
      const float* xp = x + ((size_t)b * CH + ic0 + iloc) * (HW * HW) + rc * HW;
      va[j] = (rowok && lane >= 1) ? xp[lane - 1] : 0.f;  // halo col c=lane   (img col lane-1)
      vb[j] = rowok ? xp[lane + 31] : 0.f;                // col c=lane+32     (img col lane+31)
      vc[j] = (rowok && lane == 0) ? xp[63] : 0.f;        // col c=64 (img 63); c=65 (img 64)=0
    }
  };
  // ---- modulate, convert to bf16, store channel-contiguous into LDS ----
  auto fillStore = [&](int s, __bf16* buf) {
    const int ic0 = (s & 15) << 5;
    #pragma unroll
    for (int j = 0; j < 4; ++j) {
      const int iloc = wv * 4 + j;
      const float sv = sstyle[ic0 + iloc];
      buf[lane * LSTR + iloc]        = (__bf16)(va[j] * sv);
      buf[(lane + 32) * LSTR + iloc] = (__bf16)(vb[j] * sv);
      if (lane < 2) buf[(lane + 64) * LSTR + iloc] = (__bf16)(vc[j] * sv);
    }
  };
  // ---- A fragments for the 3 kw taps of step s ----
  auto loadA = [&](int s, v16bf* af) {
    const __bf16* ab = apack +
        (((size_t)mtg * NKT + (s >> 4) * 48 + (s & 15)) * 32 + lane) * 16;
    af[0] = *(const v16bf*)(ab);
    af[1] = *(const v16bf*)(ab + 8192);    // +16 k-tiles (kw step)
    af[2] = *(const v16bf*)(ab + 16384);
  };
  // ---- 12 WMMAs: 3 kw taps x 4 N-tiles ----
  auto compute = [&](const __bf16* buf, const v16bf* af) {
    #pragma unroll
    for (int kw = 0; kw < 3; ++kw) {
      v16bf bf[4];
      #pragma unroll
      for (int nt = 0; nt < 4; ++nt) {
        const __bf16* bp = &buf[(nt * 16 + ncol + kw) * LSTR + kbase];
        v8bf lo = *(const v8bf*)(bp);
        v8bf hi = *(const v8bf*)(bp + 16);
        bf[nt] = __builtin_shufflevector(lo, hi,
            0, 1, 2, 3, 4, 5, 6, 7, 8, 9, 10, 11, 12, 13, 14, 15);
      }
      #pragma unroll
      for (int nt = 0; nt < 4; ++nt)
        acc[nt] = __builtin_amdgcn_wmma_f32_16x16x32_bf16(
            false, af[kw], false, bf[nt], (short)0, acc[nt], false, false);
    }
  };

  // ---- pipeline prologue ----
  __syncthreads();                 // sstyle visible
  fillLoad(0);
  loadA(0, afA);
  fillStore(0, xl0);
  __syncthreads();

  // ---- pipelined main loop, 2x manual unroll keeps both buffer sets in registers ----
  for (int s0 = 0; s0 < NSTEP; s0 += 2) {
    {   // even step: compute xl0/afA, prefetch (s0+1) into xl1/afB
      fillLoad(s0 + 1);
      loadA(s0 + 1, afB);
      compute(xl0, afA);
      fillStore(s0 + 1, xl1);
      __syncthreads();
    }
    {   // odd step: compute xl1/afB, prefetch (s0+2) into xl0/afA
      const bool pf = (s0 + 2) < NSTEP;
      if (pf) { fillLoad(s0 + 2); loadA(s0 + 2, afA); }
      compute(xl1, afB);
      if (pf) fillStore(s0 + 2, xl0);
      __syncthreads();
    }
  }

  // ---- epilogue: scale by rdenom and store f32 ----
  const int ohalf = (lane >> 4) * 8;
  #pragma unroll
  for (int r = 0; r < 8; ++r) {
    const int o = mbase + ohalf + r;
    const float rd = rdenom[(size_t)b * CH + o];
    float* orow = out + (((size_t)b * CH + o) * HW + y) * HW;
    #pragma unroll
    for (int nt = 0; nt < 4; ++nt)
      orow[nt * 16 + ncol] = acc[nt][r] * rd;
  }
}

extern "C" void kernel_launch(void* const* d_in, const int* in_sizes, int n_in,
                              void* d_out, int out_size, void* d_ws, size_t ws_size,
                              hipStream_t stream) {
  const float* x  = (const float*)d_in[0];   // (16,512,64,64)
  const float* w  = (const float*)d_in[1];   // (16,512)
  const float* cw = (const float*)d_in[2];   // (512,512,3,3)
  const float* lw = (const float*)d_in[3];   // (512,512)
  const float* lb = (const float*)d_in[4];   // (512)
  float* out = (float*)d_out;

  float* style  = (float*)d_ws;                             // 8192 f32
  float* rdenom = style + BATCH * CH;                       // 8192 f32
  float* wsq    = rdenom + BATCH * CH;                      // 262144 f32
  unsigned short* apack = (unsigned short*)(wsq + CH * CH); // 2,359,296 bf16 (~4.7MB)

  style_kernel <<<(BATCH * CH + 255) / 256, 256, 0, stream>>>(w, lw, lb, style);
  wsq_kernel   <<<(CH * CH + 255) / 256, 256, 0, stream>>>(cw, wsq);
  rdenom_kernel<<<(BATCH * CH + 255) / 256, 256, 0, stream>>>(style, wsq, rdenom);
  pack_kernel  <<<(NMT * NKT * 32 + 255) / 256, 256, 0, stream>>>(cw, apack);

  dim3 grid(HW, 4, BATCH);   // (row, Mtile-of-128, batch)
  modconv_kernel<<<grid, 256, 0, stream>>>(x, style, rdenom, apack, out);
}
